// LSTM_Model_76441827935010
// MI455X (gfx1250) — matrix operand order repaired and checked by reading, not compiled, not run on hardware
//
#include <hip/hip_runtime.h>
#include <hip/hip_bf16.h>
#include <math.h>

// ---------------------------------------------------------------------------
// LSTM + attention for MI455X (gfx1250), bf16 WMMA path.
//
//   z_t = [x_t, h_{t-1}] @ Wk + bias      -> v_wmma_f32_16x16x32_bf16,
//                                            LDS double-buffered pipeline
//   gates -> c_t, h_t                     -> VALU, fused repack of h into
//                                            next step's WMMA A-fragments
//   context = h_T @ out_weights           -> WMMA (reuses h_T fragments)
//   attention head (A=10)                 -> VALU (tiny)
// ---------------------------------------------------------------------------

typedef __attribute__((ext_vector_type(16))) __bf16 v16bf;
typedef __attribute__((ext_vector_type(8)))  float  v8f;

#define T_STEPS 100
#define BATCH   1024
#define DIN     300
#define HID     1024
#define FOURH   4096
#define KT_X    10            // x k-tiles (300 padded to 320)
#define KT_H    32            // h k-tiles (1024)
#define KT_ALL  42            // total K tiles, K_pad = 1344
#define NT_Z    256           // 4096 / 16 output n-tiles
#define MT      64            // 1024 / 16 output m-tiles
#define NT_CTX  19            // 300 padded to 304 -> 19 n-tiles

// ---- fragment layout (per CDNA5 ISA 7.12.2, wave32) ------------------------
// A (16x32 bf16): lane<16 holds K {0..7, 16..23}; lane>=16 holds K {8..15, 24..31}
//   inverse: lane_hi = bit3(k), elem j = (k&7) | ((k>=16) ? 8 : 0)
// B (32x16 bf16): lanes 0..15 hold K {0..7, 16..23}, lanes 16..31 hold K {8..15, 24..31}, as A

__device__ __forceinline__ float sigmf(float x) { return 1.0f / (1.0f + expf(-x)); }

#define WMMA_BF16(A, B, C) \
    __builtin_amdgcn_wmma_f32_16x16x32_bf16(false, (A), false, (B), (short)0, (C), false, false)

// ---------------------------------------------------------------------------
// Pack lstm_kernel [1324, 4096] f32 -> B-fragment bf16 [42][256][32][16]
// K-space: ktiles 0..9 = x rows 0..299 (zero padded to 320), 10..41 = h rows.
__global__ void wk_pack_kernel(const float* __restrict__ wk, __bf16* __restrict__ wfrag) {
    int id = blockIdx.x * 256 + threadIdx.x;
    if (id >= KT_ALL * NT_Z * 512) return;
    int j    = id & 15;
    int lane = (id >> 4) & 31;
    int rest = id >> 9;
    int nt   = rest % NT_Z;
    int kt   = rest / NT_Z;
    int kl   = (j & 7) + ((lane >= 16) ? 8 : 0) + ((j >= 8) ? 16 : 0); // B layout
    int col  = nt * 16 + (lane & 15);
    float v  = 0.0f;
    if (kt < KT_X) {
        int kx = kt * 32 + kl;
        if (kx < DIN) v = wk[(size_t)kx * FOURH + col];
    } else {
        int row = DIN + (kt - KT_X) * 32 + kl;    // < 1324
        v = wk[(size_t)row * FOURH + col];
    }
    wfrag[id] = (__bf16)v;
}

// Pack out_weights [1024, 300] -> B-fragment bf16 [32][19][32][16] (cols padded)
__global__ void ow_pack_kernel(const float* __restrict__ ow, __bf16* __restrict__ owfrag) {
    int id = blockIdx.x * 256 + threadIdx.x;
    if (id >= KT_H * NT_CTX * 512) return;
    int j    = id & 15;
    int lane = (id >> 4) & 31;
    int rest = id >> 9;
    int nt   = rest % NT_CTX;
    int kt   = rest / NT_CTX;
    int kl   = (j & 7) + ((lane >= 16) ? 8 : 0) + ((j >= 8) ? 16 : 0);
    int row  = kt * 32 + kl;                      // h index < 1024
    int col  = nt * 16 + (lane & 15);
    float v  = (col < DIN) ? ow[(size_t)row * DIN + col] : 0.0f;
    owfrag[id] = (__bf16)v;
}

// Pack x_t [1024, 300] f32 -> A-fragment region (ktiles 0..9) of Afrag
__global__ void xprep_kernel(const float* __restrict__ x_t, __bf16* __restrict__ afrag) {
    int id = blockIdx.x * 256 + threadIdx.x;      // BATCH * 320
    if (id >= BATCH * 320) return;
    int b = id / 320, k = id % 320;
    float v = (k < DIN) ? x_t[(size_t)b * DIN + k] : 0.0f;
    int kt   = k >> 5, kl = k & 31;
    int lane = (b & 15) | (((kl >> 3) & 1) << 4); // A layout inverse
    int j    = (kl & 7) | ((kl >= 16) ? 8 : 0);
    afrag[((((size_t)(b >> 4)) * KT_ALL + kt) * 32 + lane) * 16 + j] = (__bf16)v;
}

// ---------------------------------------------------------------------------
// Step GEMM: Z[1024,4096] = Afrag @ Wfrag.
// Block = 8 waves = 128x128 output tile; LDS double-buffered staging of the
// block's 8 A-tiles + 8 B-tiles per K step; each wave computes 2x4 WMMA tiles.
// Global loads for kt+1 are issued before the WMMAs of kt, so the matrix pipe
// only waits on short LDS reads, never on a full loadcnt drain.
__global__ __launch_bounds__(256) void lstm_gemm_kernel(
    const __bf16* __restrict__ afrag, const __bf16* __restrict__ wfrag,
    float* __restrict__ Z)
{
    __shared__ __bf16 As[2][8][512];   // [buf][mtile in block][lane*16 + j]
    __shared__ __bf16 Bs[2][8][512];   // [buf][ntile in block][lane*16 + j]

    const int tid   = threadIdx.x;
    const int lane  = tid & 31;
    const int wave  = tid >> 5;        // 0..7, also the staging tile this thread fills
    const int wm    = wave & 3;        // m-wave: owns mtiles wm*2, wm*2+1
    const int wn    = wave >> 2;       // n-wave: owns ntiles wn*4 .. wn*4+3

    const int mbase = blockIdx.x * 8;  // first mtile of block (grid.x = 8)
    const int nbase = blockIdx.y * 8;  // first ntile of block (grid.y = 32)

    // staging slot this thread owns (tile 'wave', its own lane)
    const __bf16* gA = afrag + (((size_t)(mbase + wave) * KT_ALL) * 32 + lane) * 16;
    const __bf16* gB = wfrag + (((size_t)(nbase + wave)) * 32 + lane) * 16;

    v8f acc[2][4] = {};

    // prologue: stage kt = 0
    v16bf ga = *(const v16bf*)gA;
    v16bf gb = *(const v16bf*)gB;
    *(v16bf*)&As[0][wave][lane * 16] = ga;
    *(v16bf*)&Bs[0][wave][lane * 16] = gb;
    __syncthreads();

    for (int kt = 0; kt < KT_ALL; ++kt) {
        const int buf = kt & 1;
        if (kt + 1 < KT_ALL) {                       // prefetch next K step
            ga = *(const v16bf*)(gA + (size_t)(kt + 1) * 512);
            gb = *(const v16bf*)(gB + (size_t)(kt + 1) * NT_Z * 512);
        }
        v16bf a0 = *(const v16bf*)&As[buf][wm * 2 + 0][lane * 16];
        v16bf a1 = *(const v16bf*)&As[buf][wm * 2 + 1][lane * 16];
#pragma unroll
        for (int i = 0; i < 4; ++i) {
            v16bf b = *(const v16bf*)&Bs[buf][wn * 4 + i][lane * 16];
            acc[0][i] = WMMA_BF16(a0, b, acc[0][i]);
            acc[1][i] = WMMA_BF16(a1, b, acc[1][i]);
        }
        if (kt + 1 < KT_ALL) {                       // fill other buffer
            *(v16bf*)&As[buf ^ 1][wave][lane * 16] = ga;
            *(v16bf*)&Bs[buf ^ 1][wave][lane * 16] = gb;
        }
        __syncthreads();
    }

    // C layout: VGPR r -> M = r + (lane>=16 ? 8 : 0), N = lane & 15
    const int hi8  = ((lane >> 4) ? 8 : 0);
    const int ncol = lane & 15;
#pragma unroll
    for (int mi = 0; mi < 2; ++mi) {
        const int row0 = (mbase + wm * 2 + mi) * 16 + hi8;
#pragma unroll
        for (int i = 0; i < 4; ++i) {
            float* zp = Z + (size_t)row0 * FOURH + (nbase + wn * 4 + i) * 16 + ncol;
#pragma unroll
            for (int r = 0; r < 8; ++r)
                zp[(size_t)r * FOURH] = acc[mi][i][r];
        }
    }
}

// Context GEMM: ctx[1024,300] = h_T @ out_weights, one wave per 16x16 tile.
__global__ __launch_bounds__(32) void ctx_gemm_kernel(
    const __bf16* __restrict__ afrag, const __bf16* __restrict__ owfrag,
    float* __restrict__ ctx)
{
    const int lane  = threadIdx.x & 31;
    const int mtile = blockIdx.x;                 // 0..63
    const int ntile = blockIdx.y;                 // 0..18
    v8f acc = {};
    const __bf16* ap = afrag + (((size_t)mtile * KT_ALL + KT_X) * 32 + lane) * 16;
    const __bf16* bp = owfrag + (((size_t)ntile) * 32 + lane) * 16;
    for (int kt = 0; kt < KT_H; ++kt) {
        v16bf a = *(const v16bf*)(ap + (size_t)kt * 512);
        v16bf b = *(const v16bf*)(bp + (size_t)kt * NT_CTX * 512);
        acc = WMMA_BF16(a, b, acc);
    }
    const int n = ntile * 16 + (lane & 15);
    if (n < DIN) {
        const int mbase = mtile * 16 + ((lane >> 4) ? 8 : 0);
#pragma unroll
        for (int r = 0; r < 8; ++r)
            ctx[(size_t)(mbase + r) * DIN + n] = acc[r];
    }
}

// ---------------------------------------------------------------------------
// Gates + state update; fuses bf16 repack of h into next step's A fragments
// and bf16 history write for attention.
__global__ __launch_bounds__(256) void gate_kernel(
    const float* __restrict__ Z, const float* __restrict__ bias,
    float* __restrict__ c, __bf16* __restrict__ hist_t, __bf16* __restrict__ afrag)
{
    int id = blockIdx.x * 256 + threadIdx.x;      // BATCH*HID
    int b  = id >> 10, hx = id & 1023;
    const float* zr = Z + (size_t)b * FOURH;
    float zi = zr[hx]        + bias[hx];
    float zj = zr[1024 + hx] + bias[1024 + hx];
    float zf = zr[2048 + hx] + bias[2048 + hx];
    float zo = zr[3072 + hx] + bias[3072 + hx];
    float cp = c[id];
    float cn = sigmf(zf + 1.0f) * cp + sigmf(zi) * tanhf(zj);
    float h  = sigmf(zo) * tanhf(cn);
    c[id] = cn;
    hist_t[id] = (__bf16)h;
    int kt   = KT_X + (hx >> 5), kl = hx & 31;
    int lane = (b & 15) | (((kl >> 3) & 1) << 4);
    int j    = (kl & 7) | ((kl >= 16) ? 8 : 0);
    afrag[((((size_t)(b >> 4)) * KT_ALL + kt) * 32 + lane) * 16 + j] = (__bf16)h;
}

// ---------------------------------------------------------------------------
// Attention head (A = 10): vu[t,b] = u . tanh(h W_omega + b_omega)
__global__ void vu_kernel(const __bf16* __restrict__ hist,
                          const float* __restrict__ wom, const float* __restrict__ bom,
                          const float* __restrict__ uom, float* __restrict__ vu)
{
    int id = blockIdx.x * 256 + threadIdx.x;      // T*B
    if (id >= T_STEPS * BATCH) return;
    const __bf16* hp = hist + (size_t)id * HID;
    float acc[10];
#pragma unroll
    for (int a = 0; a < 10; ++a) acc[a] = bom[a];
    for (int h = 0; h < HID; ++h) {
        float hv = (float)hp[h];
#pragma unroll
        for (int a = 0; a < 10; ++a) acc[a] += hv * wom[h * 10 + a];
    }
    float s = 0.0f;
#pragma unroll
    for (int a = 0; a < 10; ++a) s += uom[a] * tanhf(acc[a]);
    vu[id] = s;
}

__global__ void softmax_kernel(const float* __restrict__ vu, float* __restrict__ alphas) {
    int b = blockIdx.x * 256 + threadIdx.x;
    if (b >= BATCH) return;
    float m = -1e30f;
    for (int t = 0; t < T_STEPS; ++t) m = fmaxf(m, vu[t * BATCH + b]);
    float s = 0.0f;
    for (int t = 0; t < T_STEPS; ++t) {
        float e = expf(vu[t * BATCH + b] - m);
        alphas[t * BATCH + b] = e;
        s += e;
    }
    float inv = 1.0f / s;
    for (int t = 0; t < T_STEPS; ++t) alphas[t * BATCH + b] *= inv;
}

__global__ void att_kernel(const __bf16* __restrict__ hist,
                           const float* __restrict__ alphas, float* __restrict__ att)
{
    int id = blockIdx.x * 256 + threadIdx.x;      // BATCH*HID
    int b = id >> 10, hx = id & 1023;
    float s = 0.0f;
    for (int t = 0; t < T_STEPS; ++t)
        s += alphas[t * BATCH + b] * (float)hist[(((size_t)t * BATCH) + b) * HID + hx];
    att[id] = s;
}

__global__ void yhat_kernel(const float* __restrict__ att, const float* __restrict__ Wf,
                            const float* __restrict__ bsc, float* __restrict__ y)
{
    int b = blockIdx.x * 256 + threadIdx.x;
    if (b >= BATCH) return;
    float s = bsc[0];
    const float* ar = att + (size_t)b * HID;
    for (int h = 0; h < HID; ++h) s += ar[h] * Wf[h];
    y[b] = s;
}

// ---------------------------------------------------------------------------
extern "C" void kernel_launch(void* const* d_in, const int* in_sizes, int n_in,
                              void* d_out, int out_size, void* d_ws, size_t ws_size,
                              hipStream_t stream) {
    (void)in_sizes; (void)n_in; (void)out_size; (void)ws_size;
    const float* inputs = (const float*)d_in[0];   // [T,B,D]
    const float* wk     = (const float*)d_in[1];   // [D+H, 4H]
    const float* bias   = (const float*)d_in[2];   // [4H]
    const float* ow     = (const float*)d_in[3];   // [H, D]
    const float* wom    = (const float*)d_in[4];   // [H, A]
    const float* bom    = (const float*)d_in[5];   // [A]
    const float* uom    = (const float*)d_in[6];   // [A]
    const float* Wf     = (const float*)d_in[7];   // [H, 1]
    const float* bsc    = (const float*)d_in[8];   // [1]
    float* out = (float*)d_out;                    // ctx[1024*300] ++ y[1024]

    char* ws = (char*)d_ws;
    size_t off = 0;
    auto carve = [&](size_t bytes) -> char* {
        char* p = ws + off;
        off = (off + bytes + 255) & ~(size_t)255;
        return p;
    };
    float*  Z      = (float*)carve((size_t)BATCH * FOURH * 4);          // 16 MB
    float*  Cst    = (float*)carve((size_t)BATCH * HID * 4);            //  4 MB
    __bf16* Afrag  = (__bf16*)carve((size_t)MT * KT_ALL * 512 * 2);     //  2.75 MB
    __bf16* Wfrag  = (__bf16*)carve((size_t)KT_ALL * NT_Z * 512 * 2);   // 11 MB
    __bf16* OWfrag = (__bf16*)carve((size_t)KT_H * NT_CTX * 512 * 2);   //  0.6 MB
    __bf16* hist   = (__bf16*)carve((size_t)T_STEPS * BATCH * HID * 2); // 210 MB
    float*  vu     = (float*)carve((size_t)T_STEPS * BATCH * 4);
    float*  alphas = (float*)carve((size_t)T_STEPS * BATCH * 4);
    float*  att    = (float*)carve((size_t)BATCH * HID * 4);

    // zero initial c and h-fragments (t=0 uses h=0)
    hipMemsetAsync(Cst, 0, (size_t)BATCH * HID * 4, stream);
    hipMemsetAsync(Afrag, 0, (size_t)MT * KT_ALL * 512 * 2, stream);

    // one-time weight packs
    wk_pack_kernel<<<(KT_ALL * NT_Z * 512 + 255) / 256, 256, 0, stream>>>(wk, Wfrag);
    ow_pack_kernel<<<(KT_H * NT_CTX * 512 + 255) / 256, 256, 0, stream>>>(ow, OWfrag);

    // recurrent loop
    for (int t = 0; t < T_STEPS; ++t) {
        xprep_kernel<<<(BATCH * 320 + 255) / 256, 256, 0, stream>>>(
            inputs + (size_t)t * BATCH * DIN, Afrag);
        lstm_gemm_kernel<<<dim3(8, 32), 256, 0, stream>>>(Afrag, Wfrag, Z);
        gate_kernel<<<(BATCH * HID) / 256, 256, 0, stream>>>(
            Z, bias, Cst, hist + (size_t)t * BATCH * HID, Afrag);
    }

    // context_vector = h_T @ out_weights (h_T fragments still live in Afrag)
    ctx_gemm_kernel<<<dim3(MT, NT_CTX), 32, 0, stream>>>(Afrag, OWfrag, out);

    // attention head
    vu_kernel<<<(T_STEPS * BATCH + 255) / 256, 256, 0, stream>>>(hist, wom, bom, uom, vu);
    softmax_kernel<<<(BATCH + 255) / 256, 256, 0, stream>>>(vu, alphas);
    att_kernel<<<(BATCH * HID) / 256, 256, 0, stream>>>(hist, alphas, att);
    yhat_kernel<<<(BATCH + 255) / 256, 256, 0, stream>>>(att, Wf, bsc, out + BATCH * DIN);
}
